// Dominant_32590211842242
// MI455X (gfx1250) — matrix operand order repaired and checked by reading, compiled, not run on hardware
//
#include <hip/hip_runtime.h>
#include <hip/hip_bf16.h>

#define NN 16384
#define EE 524288
#define NFEAT 256
#define NHID 128

typedef __attribute__((ext_vector_type(2)))  float  v2f;
typedef __attribute__((ext_vector_type(8)))  float  v8f;
typedef __attribute__((ext_vector_type(8)))  __bf16 v8bf;
typedef __attribute__((ext_vector_type(16))) __bf16 v16bf;

// ---------------- degree / norm ----------------
__global__ void k_count(const int* __restrict__ ei, int* __restrict__ cnt) {
    int e = blockIdx.x * blockDim.x + threadIdx.x;
    if (e < EE) atomicAdd(&cnt[ei[EE + e]], 1);   // dst = ei[1][e]
}

__global__ void k_dinv(const int* __restrict__ cnt, float* __restrict__ dinv) {
    int n = blockIdx.x * blockDim.x + threadIdx.x;
    if (n < NN) dinv[n] = rsqrtf((float)cnt[n] + 1.0f);
}

// single-block exclusive scan of N=16384 counts -> rowptr (+ cursor copy)
__global__ void k_scan(const int* __restrict__ cnt, int* __restrict__ rowptr,
                       int* __restrict__ cursor) {
    __shared__ int part[256];
    int t = threadIdx.x;
    int base = t * 64;
    int s = 0;
    for (int i = 0; i < 64; ++i) s += cnt[base + i];
    part[t] = s;
    __syncthreads();
    if (t == 0) {
        int acc = 0;
        for (int i = 0; i < 256; ++i) { int v = part[i]; part[i] = acc; acc += v; }
    }
    __syncthreads();
    int run = part[t];
    for (int i = 0; i < 64; ++i) {
        int v = cnt[base + i];
        rowptr[base + i] = run;
        cursor[base + i] = run;
        run += v;
    }
    if (t == 255) rowptr[NN] = run;
}

__global__ void k_fill(const int* __restrict__ ei, int* __restrict__ cursor,
                       int* __restrict__ csr) {
    int e = blockIdx.x * blockDim.x + threadIdx.x;
    if (e < EE) {
        int dst = ei[EE + e];
        int pos = atomicAdd(&cursor[dst], 1);
        csr[pos] = ei[e];                          // src = ei[0][e]
    }
}

// deterministic gather order: sort each row's src indices (avg row len = 32)
__global__ void k_sortrows(const int* __restrict__ rowptr, int* __restrict__ csr) {
    int n = blockIdx.x * blockDim.x + threadIdx.x;
    if (n >= NN) return;
    int beg = rowptr[n], end = rowptr[n + 1];
    for (int i = beg + 1; i < end; ++i) {
        int v = csr[i];
        int j = i - 1;
        while (j >= beg && csr[j] > v) { csr[j + 1] = csr[j]; --j; }
        csr[j + 1] = v;
    }
}

// ---------------- fp32 GEMM (h = A @ W) via V_WMMA_F32_16X16X4_F32 ----------------
__global__ void k_gemm_f32(const float* __restrict__ A, const float* __restrict__ W,
                           float* __restrict__ out, int M, int K, int Fout) {
    int wave = (blockIdx.x * blockDim.x + threadIdx.x) >> 5;
    int lane = threadIdx.x & 31;
    int tilesN = Fout >> 4;
    int tm = wave / tilesN, tn = wave % tilesN;
    if (tm * 16 >= M) return;
    int hl = lane >> 4;           // half-wave select
    int mr = lane & 15;
    v8f acc = {};
#if __has_builtin(__builtin_amdgcn_wmma_f32_16x16x4_f32)
    const float* arow = A + (size_t)(tm * 16 + mr) * K;
    const float* bcol = W + (tn * 16 + mr);
    for (int k = 0; k < K; k += 4) {
        int ka = k + hl * 2;
        v2f a, b;
        a.x = arow[ka];
        a.y = arow[ka + 1];
        b.x = bcol[(size_t)ka * Fout];
        b.y = bcol[(size_t)(ka + 1) * Fout];
        acc = __builtin_amdgcn_wmma_f32_16x16x4_f32(
            false, a, false, b, (short)0, acc, false, false);
    }
#else
    // scalar fallback, identical D layout
    for (int j = 0; j < 8; ++j) {
        const float* arow = A + (size_t)(tm * 16 + j + 8 * hl) * K;
        float s = 0.f;
        for (int k = 0; k < K; ++k) s += arow[k] * W[(size_t)k * Fout + tn * 16 + mr];
        acc[j] = s;
    }
#endif
#pragma unroll
    for (int j = 0; j < 8; ++j)
        out[(size_t)(tm * 16 + j + 8 * hl) * Fout + tn * 16 + mr] = acc[j];
}

// ---------------- GCN aggregation + bias + ReLU ----------------
// out[n][t] = relu( dinv[n]*sum_{e in row n} dinv[src]*h[src][t] + dinv[n]^2*h[n][t] + b[t] )
#define CHUNK 128
__global__ void k_agg(const float* __restrict__ h, const float* __restrict__ dinv,
                      const int* __restrict__ rowptr, const int* __restrict__ csr,
                      const float* __restrict__ bias, float* __restrict__ out, int F) {
    __shared__ int   s_idx[CHUNK];
    __shared__ float s_w[CHUNK];
    int n = blockIdx.x;
    int t = threadIdx.x;
    int beg = rowptr[n], end = rowptr[n + 1];
    float acc = 0.f;
    for (int c = beg; c < end; c += CHUNK) {
        int m = end - c; if (m > CHUNK) m = CHUNK;
        __syncthreads();
        for (int i = t; i < m; i += blockDim.x) {
            int s = csr[c + i];
            s_idx[i] = s;
            s_w[i]   = dinv[s];
        }
        __syncthreads();
        for (int i = 0; i < m; ++i)
            acc += s_w[i] * h[(size_t)s_idx[i] * F + t];
    }
    float dn = dinv[n];
    float v = dn * acc + dn * dn * h[(size_t)n * F + t] + bias[t];
    out[(size_t)n * F + t] = fmaxf(v, 0.f);
}

// ---------------- f32 -> bf16 (RNE) ----------------
__global__ void k_tobf16(const float* __restrict__ in, unsigned short* __restrict__ out,
                         int n) {
    int i = blockIdx.x * blockDim.x + threadIdx.x;
    if (i < n) {
        unsigned u = __builtin_bit_cast(unsigned, in[i]);
        unsigned r = u + 0x7fffu + ((u >> 16) & 1u);
        out[i] = (unsigned short)(r >> 16);
    }
}

// ---------------- A_hat = s @ s^T, LDS-tiled, V_WMMA_F32_16X16X32_BF16 ----------------
// Block: 256 threads (8 waves) computes a 128x128 output block.
// Stage s-rows for A (bm) and B (bn) in LDS once: 64 KB global traffic per 64 KB output.
// LDS rows padded to 272 B so 16-lane operand reads shift 4 banks/row (conflict-free).
#define BT 128
#define LSTR 136   // padded LDS row stride in bf16 elements (272 bytes)

__global__ __launch_bounds__(256) void k_ahat(const unsigned short* __restrict__ Sb,
                                              float* __restrict__ out) {
    __shared__ unsigned short sA[BT * LSTR];
    __shared__ unsigned short sB[BT * LSTR];
    int bm = blockIdx.x / (NN / BT);
    int bn = blockIdx.x % (NN / BT);
    int t = threadIdx.x;

    // cooperative stage: 128 rows x 256 B each; thread t loads half a row (8 x 16 B)
    {
        int row = t >> 1;
        int hf  = t & 1;
        const uint4* gA = (const uint4*)(Sb + (size_t)(bm * BT + row) * NHID + hf * 64);
        const uint4* gB = (const uint4*)(Sb + (size_t)(bn * BT + row) * NHID + hf * 64);
        uint4* lA = (uint4*)(sA + row * LSTR + hf * 64);
        uint4* lB = (uint4*)(sB + row * LSTR + hf * 64);
#pragma unroll
        for (int i = 0; i < 8; ++i) { lA[i] = gA[i]; lB[i] = gB[i]; }
    }
    __syncthreads();

    int wv   = t >> 5;            // wave 0..7
    int lane = t & 31;
    int hl = lane >> 4;
    int mr = lane & 15;
    int wm = (wv >> 1) * 32;      // wave M offset: 0,32,64,96
    int wn = (wv & 1) * 64;       // wave N offset: 0,64

    const __bf16* lsA = (const __bf16*)sA;
    const __bf16* lsB = (const __bf16*)sB;

    v8f acc[2][4] = {};
#pragma unroll
    for (int k = 0; k < NHID; k += 32) {
        v16bf a[2], b[4];
#pragma unroll
        for (int mi = 0; mi < 2; ++mi) {
            const __bf16* ar = lsA + (size_t)(wm + mi * 16 + mr) * LSTR;
            v8bf lo = *(const v8bf*)(ar + k + 8 * hl);
            v8bf hi = *(const v8bf*)(ar + k + 16 + 8 * hl);
            a[mi] = __builtin_shufflevector(lo, hi,
                        0, 1, 2, 3, 4, 5, 6, 7, 8, 9, 10, 11, 12, 13, 14, 15);
        }
#pragma unroll
        for (int ni = 0; ni < 4; ++ni) {
            const __bf16* br = lsB + (size_t)(wn + ni * 16 + mr) * LSTR;
            v8bf blo = *(const v8bf*)(br + k + 8 * hl);
            v8bf bhi = *(const v8bf*)(br + k + 16 + 8 * hl);
            b[ni] = __builtin_shufflevector(blo, bhi,
                        0, 1, 2, 3, 4, 5, 6, 7, 8, 9, 10, 11, 12, 13, 14, 15);
        }
#pragma unroll
        for (int mi = 0; mi < 2; ++mi)
#pragma unroll
            for (int ni = 0; ni < 4; ++ni)
                acc[mi][ni] = __builtin_amdgcn_wmma_f32_16x16x32_bf16(
                    false, a[mi], false, b[ni], (short)0, acc[mi][ni], false, false);
    }

#pragma unroll
    for (int mi = 0; mi < 2; ++mi) {
#pragma unroll
        for (int j = 0; j < 8; ++j) {
            size_t row = (size_t)(bm * BT + wm + mi * 16 + j + 8 * hl);
            float* orow = out + row * NN + bn * BT + wn + mr;
#pragma unroll
            for (int ni = 0; ni < 4; ++ni)
                orow[ni * 16] = acc[mi][ni][j];
        }
    }
}

// ---------------- host-side launch ----------------
extern "C" void kernel_launch(void* const* d_in, const int* in_sizes, int n_in,
                              void* d_out, int out_size, void* d_ws, size_t ws_size,
                              hipStream_t stream) {
    const float* x   = (const float*)d_in[0];
    const int*   ei  = (const int*)d_in[1];
    const float* We1 = (const float*)d_in[2];  const float* be1 = (const float*)d_in[3];
    const float* We2 = (const float*)d_in[4];  const float* be2 = (const float*)d_in[5];
    const float* Wa1 = (const float*)d_in[6];  const float* ba1 = (const float*)d_in[7];
    const float* Wa2 = (const float*)d_in[8];  const float* ba2 = (const float*)d_in[9];
    const float* Ws1 = (const float*)d_in[10]; const float* bs1 = (const float*)d_in[11];

    char* ws = (char*)d_ws;
    int*   cnt     = (int*)(ws + 0);                       //  64 KB
    float* dinv    = (float*)(ws + (1u << 16));            //  64 KB
    int*   rowptr  = (int*)(ws + (2u << 16));              //  64 KB + 4
    int*   cursor  = (int*)(ws + (2u << 16) + 69632);      //  64 KB
    int*   csr     = (int*)(ws + (2u << 16) + 69632 + 65536);   // 2 MB
    float* bufH    = (float*)(ws + (4u << 20));            // 16 MB  (pre-agg h, up to N*256)
    float* buf1    = (float*)(ws + (20u << 20));           //  8 MB  (z1 / a / s)
    float* bufZ    = (float*)(ws + (28u << 20));           //  8 MB  (z)
    unsigned short* sb16 = (unsigned short*)(ws + (36u << 20)); // 4 MB

    float* A_hat = (float*)d_out;
    float* X_hat = (float*)d_out + (size_t)NN * NN;

    // --- normalization + CSR-by-dst (deterministic) ---
    hipMemsetAsync(cnt, 0, NN * sizeof(int), stream);
    k_count<<<EE / 256, 256, 0, stream>>>(ei, cnt);
    k_dinv<<<NN / 256, 256, 0, stream>>>(cnt, dinv);
    k_scan<<<1, 256, 0, stream>>>(cnt, rowptr, cursor);
    k_fill<<<EE / 256, 256, 0, stream>>>(ei, cursor, csr);
    k_sortrows<<<NN / 256, 256, 0, stream>>>(rowptr, csr);

    // --- layer 1: z1 = relu(gcn(x, We1, be1)) ---
    {
        int tiles = (NN / 16) * (NHID / 16);               // 8192
        k_gemm_f32<<<tiles * 32 / 128, 128, 0, stream>>>(x, We1, bufH, NN, NFEAT, NHID);
        k_agg<<<NN, NHID, 0, stream>>>(bufH, dinv, rowptr, csr, be1, buf1, NHID);
    }
    // --- layer 2: z = relu(gcn(z1, We2, be2)) ---
    {
        int tiles = (NN / 16) * (NHID / 16);
        k_gemm_f32<<<tiles * 32 / 128, 128, 0, stream>>>(buf1, We2, bufH, NN, NHID, NHID);
        k_agg<<<NN, NHID, 0, stream>>>(bufH, dinv, rowptr, csr, be2, bufZ, NHID);
    }
    // --- attr decoder 1: a = relu(gcn(z, Wa1, ba1)) ---
    {
        int tiles = (NN / 16) * (NHID / 16);
        k_gemm_f32<<<tiles * 32 / 128, 128, 0, stream>>>(bufZ, Wa1, bufH, NN, NHID, NHID);
        k_agg<<<NN, NHID, 0, stream>>>(bufH, dinv, rowptr, csr, ba1, buf1, NHID);
    }
    // --- attr decoder 2: X_hat = relu(gcn(a, Wa2, ba2)) ---
    {
        int tiles = (NN / 16) * (NFEAT / 16);              // 16384
        k_gemm_f32<<<tiles * 32 / 128, 128, 0, stream>>>(buf1, Wa2, bufH, NN, NHID, NFEAT);
        k_agg<<<NN, NFEAT, 0, stream>>>(bufH, dinv, rowptr, csr, ba2, X_hat, NFEAT);
    }
    // --- struct decoder: s = relu(gcn(z, Ws1, bs1)) ---
    {
        int tiles = (NN / 16) * (NHID / 16);
        k_gemm_f32<<<tiles * 32 / 128, 128, 0, stream>>>(bufZ, Ws1, bufH, NN, NHID, NHID);
        k_agg<<<NN, NHID, 0, stream>>>(bufH, dinv, rowptr, csr, bs1, buf1, NHID);
    }
    // --- A_hat = s @ s^T (bf16 WMMA, f32 accumulate, LDS-tiled) ---
    k_tobf16<<<(NN * NHID) / 256, 256, 0, stream>>>(buf1, sb16, NN * NHID);
    {
        int blocks = (NN / BT) * (NN / BT);                // 16384 blocks of 8 waves
        k_ahat<<<blocks, 256, 0, stream>>>(sb16, A_hat);
    }
}
